// MambaSSM_40192303956182
// MI455X (gfx1250) — matrix operand order repaired and checked by reading, not compiled, not run on hardware
//
#include <hip/hip_runtime.h>
#include <hip/hip_bf16.h>
#include <cstdint>

typedef __bf16 bf16_t;
typedef __attribute__((ext_vector_type(16))) __bf16 v16bf;
typedef __attribute__((ext_vector_type(8)))  __bf16 v8bf;
typedef __attribute__((ext_vector_type(4)))  __bf16 v4bf;
typedef __attribute__((ext_vector_type(8)))  float  v8f;
typedef __attribute__((ext_vector_type(4)))  unsigned int u32x4;
typedef __attribute__((ext_vector_type(8)))  int i32x8;
typedef __attribute__((ext_vector_type(4)))  int i32x4;

constexpr int D_MODEL = 1024;
constexpr int D_INNER = 2048;
constexpr int BATCH   = 4;
constexpr int SEQ     = 4096;
constexpr int TOKENS  = BATCH * SEQ;   // 16384

__device__ __forceinline__ float silu_f(float v) { return v / (1.f + __expf(-v)); }

#if __has_builtin(__builtin_amdgcn_tensor_load_to_lds) && __has_builtin(__builtin_amdgcn_s_wait_tensorcnt)
#define HAVE_TDM 1
#endif

// ---------------- f32 -> bf16 bulk conversion ----------------
__global__ __launch_bounds__(256)
void cvt_f32_bf16(const float* __restrict__ in, bf16_t* __restrict__ out, int n4)
{
    const int i = blockIdx.x * 256 + threadIdx.x;
    if (i < n4) {
        float4 v = ((const float4*)in)[i];
        v4bf p; p[0]=(bf16_t)v.x; p[1]=(bf16_t)v.y; p[2]=(bf16_t)v.z; p[3]=(bf16_t)v.w;
        ((v4bf*)out)[i] = p;
    }
}

// ---------------- WMMA bf16 GEMM:  C(M,N) = A(M,K) * W(N,K)^T ----------------
constexpr int TM = 128, TN = 128, TK = 32;
constexpr int LDSS = TK + 8;   // 40 elements = 80B row stride (16B aligned, conflict-free)

union FragU { v8bf h[2]; v16bf v; };

#ifdef HAVE_TDM
// Issue one TDM 2D-tile load: 32 x 128 bf16 tile, row stride = stride_elems,
// LDS destination padded 4 DWORDs every 16 DWORDs (=> 40-element row stride).
__device__ __forceinline__ void tdm_load_tile(unsigned lds_off, const bf16_t* g,
                                              unsigned stride_elems)
{
    const unsigned long long ga = (unsigned long long)(uintptr_t)g;
    u32x4 g0 = {0u, 0u, 0u, 0u};
    g0[0] = 1u;                                         // count = 1 (valid descriptor)
    g0[1] = lds_off;                                    // lds_addr (bytes)
    g0[2] = (unsigned)ga;                               // global_addr[31:0]
    g0[3] = ((unsigned)(ga >> 32) & 0x1ffffffu) | (2u << 30);  // ga[56:32] | type=2
    i32x8 g1 = {0, 0, 0, 0, 0, 0, 0, 0};
    // data_size=2B, pad_enable, pad_interval=16 DWORDs (code 3), pad_amount=4 DWORDs (code 3)
    g1[0] = (int)((1u << 16) | (1u << 20) | (3u << 22) | (3u << 25));
    g1[1] = (int)((32u & 0xffffu) << 16);               // tensor_dim0 = 32 (lo16 @ bit48)
    g1[2] = (int)(((32u >> 16) & 0xffffu) | ((128u & 0xffffu) << 16)); // dim0 hi | tensor_dim1 lo
    g1[3] = (int)(((128u >> 16) & 0xffffu) | (32u << 16));             // dim1 hi | tile_dim0=32
    g1[4] = (int)(128u);                                // tile_dim1 = 128, tile_dim2 = 0
    g1[5] = (int)stride_elems;                          // tensor_dim0_stride[31:0]
    const i32x4 gz = {0, 0, 0, 0};
#if defined(__clang_major__) && (__clang_major__ >= 23)
    const i32x8 gz8 = {0, 0, 0, 0, 0, 0, 0, 0};
    __builtin_amdgcn_tensor_load_to_lds(g0, g1, gz, gz, gz8, 0);
#else
    __builtin_amdgcn_tensor_load_to_lds(g0, g1, gz, gz, 0);
#endif
}
#endif

__device__ __forceinline__ void gemm_tile_compute(const bf16_t (*lA)[LDSS],
                                                  const bf16_t (*lW)[LDSS],
                                                  int wm, int wn, int half, int l16,
                                                  v8f acc[2][4])
{
    // A fragments (16x32 bf16): lanes 0-15: row=l16, K {0..7,16..23};
    //                           lanes 16-31: row=l16, K {8..15,24..31}
    FragU af[2];
    #pragma unroll
    for (int mi = 0; mi < 2; ++mi) {
        const int r   = wm * 32 + mi * 16 + l16;
        const int klo = half * 8;
        af[mi].h[0] = *(const v8bf*)&lA[r][klo];
        af[mi].h[1] = *(const v8bf*)&lA[r][klo + 16];
    }
    // B fragments (32x16 bf16, LDS tile already holds B^T = W(N,K)):
    //   lanes 0-15: col=l16, K {0..7,16..23};  lanes 16-31: col=l16, K {8..15,24..31}
    FragU bfr[4];
    #pragma unroll
    for (int ni = 0; ni < 4; ++ni) {
        const int n   = wn * 64 + ni * 16 + l16;
        const int klo = half * 8;
        bfr[ni].h[0] = *(const v8bf*)&lW[n][klo];
        bfr[ni].h[1] = *(const v8bf*)&lW[n][klo + 16];
    }
    #pragma unroll
    for (int mi = 0; mi < 2; ++mi)
        #pragma unroll
        for (int ni = 0; ni < 4; ++ni)
            acc[mi][ni] = __builtin_amdgcn_wmma_f32_16x16x32_bf16(
                false, af[mi].v, false, bfr[ni].v, (short)0, acc[mi][ni], false, false);
}

__global__ __launch_bounds__(256)
void mamba_gemm_bf16(const bf16_t* __restrict__ A, const bf16_t* __restrict__ W,
                     int M, int N, int K,
                     float* __restrict__ out0, float* __restrict__ out1, int mode)
{
    __shared__ bf16_t lA[2][TM][LDSS];
    __shared__ bf16_t lW[2][TN][LDSS];

    const int tid  = threadIdx.x;
    const int lane = tid & 31;
    const int wid  = tid >> 5;
    const int wm   = wid & 3;          // 4 waves along M (32 rows each)
    const int wn   = wid >> 2;         // 2 waves along N (64 cols each)
    const int half = lane >> 4;
    const int l16  = lane & 15;

    const int m_base = blockIdx.y * TM;
    const int n_base = blockIdx.x * TN;

    v8f acc[2][4] = {};

#ifdef HAVE_TDM
    // ---- TDM double-buffered pipeline: wave 0 drives the Tensor Data Mover ----
    const unsigned ldsA0 = (unsigned)(uintptr_t)&lA[0][0][0];
    const unsigned ldsA1 = (unsigned)(uintptr_t)&lA[1][0][0];
    const unsigned ldsW0 = (unsigned)(uintptr_t)&lW[0][0][0];
    const unsigned ldsW1 = (unsigned)(uintptr_t)&lW[1][0][0];

    if (wid == 0) {   // prologue: buffer 0
        tdm_load_tile(ldsA0, A + (size_t)m_base * K, (unsigned)K);
        tdm_load_tile(ldsW0, W + (size_t)n_base * K, (unsigned)K);
    }
    int cur = 0;
    for (int kt = 0; kt < K; kt += TK) {
        const bool has_next = (kt + TK) < K;
        if (wid == 0) {
            if (has_next) {
                tdm_load_tile(cur ? ldsA0 : ldsA1, A + (size_t)m_base * K + kt + TK, (unsigned)K);
                tdm_load_tile(cur ? ldsW0 : ldsW1, W + (size_t)n_base * K + kt + TK, (unsigned)K);
                __builtin_amdgcn_s_wait_tensorcnt(2);   // retire current buffer's pair
            } else {
                __builtin_amdgcn_s_wait_tensorcnt(0);   // last tile: drain everything
            }
        }
        __syncthreads();                         // current buffer ready for everyone
        gemm_tile_compute(lA[cur], lW[cur], wm, wn, half, l16, acc);
        __syncthreads();                         // all reads done before next overwrite
        cur ^= 1;
    }
#else
    // ---- fallback: manual bf16 tile copies (loads batched before LDS stores) ----
    const int q  = tid & 3;            // 4 x 16B chunks per 64B row
    const int r0 = tid >> 2;           // 64 rows per pass, 2 passes
    for (int kt = 0; kt < K; kt += TK) {
        uint4 ta[2], tw[2];
        #pragma unroll
        for (int p = 0; p < 2; ++p) {
            const int r = r0 + p * 64;
            ta[p] = *(const uint4*)(A + (size_t)(m_base + r) * K + kt + q * 8);
            tw[p] = *(const uint4*)(W + (size_t)(n_base + r) * K + kt + q * 8);
        }
        if (kt + TK < K) {
            __builtin_prefetch(A + (size_t)(m_base + r0) * K + kt + TK + q * 8, 0, 3);
            __builtin_prefetch(W + (size_t)(n_base + r0) * K + kt + TK + q * 8, 0, 3);
        }
        #pragma unroll
        for (int p = 0; p < 2; ++p) {
            const int r = r0 + p * 64;
            *(uint4*)&lA[0][r][q * 8] = ta[p];
            *(uint4*)&lW[0][r][q * 8] = tw[p];
        }
        __syncthreads();
        gemm_tile_compute(lA[0], lW[0], wm, wn, half, l16, acc);
        __syncthreads();
    }
#endif

    // ---- epilogue: C layout: lane<16 -> N=l16, M=r ; lane>=16 -> N=l16, M=r+8 ----
    const int Nh = N >> 1;
    #pragma unroll
    for (int mi = 0; mi < 2; ++mi)
        #pragma unroll
        for (int ni = 0; ni < 4; ++ni) {
            const int n = n_base + wn * 64 + ni * 16 + l16;
            #pragma unroll
            for (int r = 0; r < 8; ++r) {
                const int m = m_base + wm * 32 + mi * 16 + half * 8 + r;
                const float v = acc[mi][ni][r];
                if (mode == 0) {   // split xz -> x_inner | z
                    if (n < Nh) out0[(size_t)m * Nh + n]        = v;
                    else        out1[(size_t)m * Nh + (n - Nh)] = v;
                } else {
                    out0[(size_t)m * N + n] = v;
                }
            }
        }
}

// ---------------- depthwise causal conv1d (k=4) + SiLU ----------------
__global__ __launch_bounds__(256)
void mamba_conv_silu(const float* __restrict__ x_inner, const float* __restrict__ conv_w,
                     const float* __restrict__ conv_b, float* __restrict__ x_conv)
{
    const size_t gid = (size_t)blockIdx.x * blockDim.x + threadIdx.x;
    const int d   = (int)(gid & (D_INNER - 1));
    const int tok = (int)(gid >> 11);
    const int l   = tok & (SEQ - 1);
    float4 w = *(const float4*)(conv_w + d * 4);
    const float wk[4] = {w.x, w.y, w.z, w.w};
    float s = conv_b[d];
    #pragma unroll
    for (int k = 0; k < 4; ++k) {
        const int li = l - 3 + k;
        if (li >= 0) s += wk[k] * x_inner[((size_t)tok - 3 + k) * D_INNER + d];
    }
    x_conv[gid] = silu_f(s);
}

// ---------------- x_proj (33 outputs) + softplus/exp postprocessing ----------------
__global__ __launch_bounds__(256)
void mamba_xproj(const float* __restrict__ x_conv, const float* __restrict__ Wx,
                 const float* __restrict__ A_log,
                 float* __restrict__ dAb, float* __restrict__ dBb, float* __restrict__ Cb)
{
    __shared__ float sred[8][34];
    const int lane  = threadIdx.x & 31;
    const int wid   = threadIdx.x >> 5;
    const int token = blockIdx.x * 8 + wid;

    float acc[33];
    #pragma unroll
    for (int e = 0; e < 33; ++e) acc[e] = 0.f;

    const float* xrow = x_conv + (size_t)token * D_INNER;
    for (int i = 0; i < D_INNER / 32; ++i) {
        const int dim = lane + (i << 5);
        const float xv = xrow[dim];
        #pragma unroll
        for (int e = 0; e < 33; ++e) acc[e] += xv * Wx[e * D_INNER + dim];
    }
    #pragma unroll
    for (int e = 0; e < 33; ++e) {
        #pragma unroll
        for (int off = 16; off > 0; off >>= 1) acc[e] += __shfl_xor(acc[e], off, 32);
    }
    if (lane == 0) {
        #pragma unroll
        for (int e = 0; e < 33; ++e) sred[wid][e] = acc[e];
    }
    __syncthreads();
    const float v0    = sred[wid][0];
    const float delta = (v0 > 20.f) ? v0 : log1pf(__expf(v0));
    if (lane < 16) {
        const float An = -__expf(A_log[lane]);               // A = -exp(A_log)
        dAb[(size_t)token * 16 + lane] = __expf(delta * An); // exp(delta*A[n])
        dBb[(size_t)token * 16 + lane] = delta * sred[wid][1 + lane];
        Cb [(size_t)token * 16 + lane] = sred[wid][17 + lane];
    }
}

// -------- sequential selective scan; x/z/coeffs LDS-staged per 64-token chunk --------
constexpr int CHUNK = 64;
constexpr int SCAN_CH = 64;   // channels per block == blockDim

__global__ __launch_bounds__(SCAN_CH)
void mamba_scan(const float* __restrict__ x_conv, const float* __restrict__ zbuf,
                const float* __restrict__ dAb, const float* __restrict__ dBb,
                const float* __restrict__ Cb, const float* __restrict__ Dp,
                bf16_t* __restrict__ ybf)
{
    __shared__ float sx[CHUNK][SCAN_CH];   // 16 KB
    __shared__ float sz[CHUNK][SCAN_CH];   // 16 KB
    __shared__ float sA[CHUNK][16];        // 4 KB
    __shared__ float sB[CHUNK][16];        // 4 KB
    __shared__ float sC[CHUNK][16];        // 4 KB

    const int tid = threadIdx.x;
    const int b   = blockIdx.x >> 5;                    // 32 blocks per batch
    const int d0  = (blockIdx.x & 31) * SCAN_CH;
    const int d   = d0 + tid;
    const float Dd = Dp[d];

    float h[16];
    #pragma unroll
    for (int n = 0; n < 16; ++n) h[n] = 0.f;

    for (int c = 0; c < SEQ / CHUNK; ++c) {
        const size_t tb = (size_t)b * SEQ + (size_t)c * CHUNK;
        __syncthreads();
        #pragma unroll 4
        for (int j = 0; j < (CHUNK * 16) / SCAN_CH; ++j) {
            const int linear = tid + j * SCAN_CH;
            ((float*)sA)[linear] = dAb[tb * 16 + linear];
            ((float*)sB)[linear] = dBb[tb * 16 + linear];
            ((float*)sC)[linear] = Cb [tb * 16 + linear];
        }
        #pragma unroll 8
        for (int t = 0; t < CHUNK; ++t) {
            sx[t][tid] = x_conv[(tb + t) * D_INNER + d0 + tid];
            sz[t][tid] = zbuf  [(tb + t) * D_INNER + d0 + tid];
        }
        __syncthreads();
        for (int t = 0; t < CHUNK; ++t) {
            const float xv = sx[t][tid];
            const float zv = sz[t][tid];
            float y = 0.f;
            #pragma unroll
            for (int n = 0; n < 16; ++n) {
                h[n] = h[n] * sA[t][n] + xv * sB[t][n];
                y += h[n] * sC[t][n];
            }
            ybf[(tb + t) * D_INNER + d] = (bf16_t)((y + xv * Dd) * silu_f(zv));
        }
    }
}

// ---------------- launcher ----------------
extern "C" void kernel_launch(void* const* d_in, const int* in_sizes, int n_in,
                              void* d_out, int out_size, void* d_ws, size_t ws_size,
                              hipStream_t stream)
{
    const float* x          = (const float*)d_in[0];
    const float* in_proj_w  = (const float*)d_in[1];
    const float* conv_w     = (const float*)d_in[2];
    const float* conv_b     = (const float*)d_in[3];
    const float* x_proj_w   = (const float*)d_in[4];
    const float* A_log      = (const float*)d_in[5];
    const float* Dp         = (const float*)d_in[6];
    const float* out_proj_w = (const float*)d_in[7];
    float* out = (float*)d_out;

    float* ws = (float*)d_ws;
    const size_t act = (size_t)TOKENS * D_INNER;
    float* x_inner = ws;
    float* zbuf    = x_inner + act;
    float* x_conv  = zbuf + act;
    float* dAb     = x_conv + act;
    float* dBb     = dAb + (size_t)TOKENS * 16;
    float* Cb      = dBb + (size_t)TOKENS * 16;
    bf16_t* xbf  = (bf16_t*)(Cb + (size_t)TOKENS * 16);
    bf16_t* w1bf = xbf  + (size_t)TOKENS * D_MODEL;
    bf16_t* w2bf = w1bf + (size_t)4096 * 1024;
    bf16_t* ybf  = w2bf + (size_t)1024 * 2048;

    // 0) one-time f32 -> bf16 conversions (activations converted once, not per tile)
    cvt_f32_bf16<<<(TOKENS * D_MODEL / 4 + 255) / 256, 256, 0, stream>>>(x, xbf, TOKENS * D_MODEL / 4);
    cvt_f32_bf16<<<(4096 * 1024 / 4 + 255) / 256, 256, 0, stream>>>(in_proj_w, w1bf, 4096 * 1024 / 4);
    cvt_f32_bf16<<<(1024 * 2048 / 4 + 255) / 256, 256, 0, stream>>>(out_proj_w, w2bf, 1024 * 2048 / 4);

    // 1) xz = x @ in_proj_w^T  (16384 x 4096, K=1024) -> split x_inner | z
    mamba_gemm_bf16<<<dim3(4096 / TN, TOKENS / TM), 256, 0, stream>>>(
        xbf, w1bf, TOKENS, 4096, D_MODEL, x_inner, zbuf, 0);

    // 2) causal depthwise conv + SiLU
    mamba_conv_silu<<<(TOKENS * D_INNER) / 256, 256, 0, stream>>>(
        x_inner, conv_w, conv_b, x_conv);

    // 3) x_proj + postprocessing -> per-token dA/dB/C
    mamba_xproj<<<TOKENS / 8, 256, 0, stream>>>(x_conv, x_proj_w, A_log, dAb, dBb, Cb);

    // 4) selective scan -> y (bf16, ready for out_proj GEMM)
    mamba_scan<<<BATCH * (D_INNER / SCAN_CH), SCAN_CH, 0, stream>>>(
        x_conv, zbuf, dAb, dBb, Cb, Dp, ybf);

    // 5) out = y @ out_proj_w^T  (16384 x 1024, K=2048)
    mamba_gemm_bf16<<<dim3(D_MODEL / TN, TOKENS / TM), 256, 0, stream>>>(
        ybf, w2bf, TOKENS, D_MODEL, D_INNER, out, nullptr, 1);
}
